// ElmanRNN_circulant_67740224192871
// MI455X (gfx1250) — matrix operand order repaired and checked by reading, not compiled, not run on hardware
//
#include <hip/hip_runtime.h>
#include <hip/hip_bf16.h>
#include <math.h>

// Problem constants
#define BB 32
#define TT 512
#define NN 512
#define HH 1024
#define OO 512
// derived tile counts (16x16x32 bf16 WMMA tiles)
#define MT1 1024   // (B*T)/16 rows for GEMM1/GEMM3
#define KT1 16     // N/32
#define CT1 64     // H/16
#define KTH 32     // H/32
#define CTH 64     // H/16
#define CTO 32     // O/16

typedef __attribute__((ext_vector_type(16))) __bf16 v16bf;
typedef __attribute__((ext_vector_type(8)))  __bf16 v8bf;
typedef __attribute__((ext_vector_type(8)))  float  v8f;
typedef __attribute__((ext_vector_type(4)))  unsigned int v4u;
typedef __attribute__((ext_vector_type(8)))  int v8i;
typedef __attribute__((ext_vector_type(4)))  int v4i;

static __device__ __forceinline__ v16bf cat16(v8bf lo, v8bf hi) {
  return __builtin_shufflevector(lo, hi, 0,1,2,3,4,5,6,7,8,9,10,11,12,13,14,15);
}

// Issue a TDM 2D tile load: 16 rows x 1024 f32, row stride TT*HH elements,
// global tile start = gaddr, LDS dest = lds_off. D# layout per CDNA5 ISA 8.3/8.4.
static __device__ __forceinline__ void tdm_load_xp_tile(unsigned lds_off,
                                                        unsigned long long gaddr) {
  v4u g0;
  g0.x = 1u;                                   // count=1, is_restore=0, gather=0
  g0.y = lds_off;                              // lds_addr (bytes)
  g0.z = (unsigned)(gaddr & 0xffffffffull);    // global_addr[31:0]
  g0.w = (unsigned)((gaddr >> 32) & 0x1ffffffull) | (2u << 30); // addr[56:32], type=2
  v8i g1;
  g1[0] = (2 << 16);                           // wg_mask=0, data_size=4B
  g1[1] = (int)(1024u << 16);                  // tensor_dim0[15:0] in [31:16]
  g1[2] = (int)(16u << 16);                    // tensor_dim0[31:16]=0 | tensor_dim1[15:0]
  g1[3] = (int)(1024u << 16);                  // tensor_dim1[31:16]=0 | tile_dim0=1024
  g1[4] = 16;                                  // tile_dim1=16, tile_dim2=0
  g1[5] = (int)(TT * HH);                      // tensor_dim0_stride[31:0]
  g1[6] = 0;                                   // stride0[47:32]=0, stride1[15:0]=0
  g1[7] = 0;                                   // stride1[47:16]=0
  v4i gz = {0, 0, 0, 0};
#if __clang_major__ >= 23
  v8i gz8 = {0, 0, 0, 0, 0, 0, 0, 0};
  __builtin_amdgcn_tensor_load_to_lds(g0, g1, gz, gz, gz8, 0);
#else
  __builtin_amdgcn_tensor_load_to_lds(g0, g1, gz, gz, 0);
#endif
}

// ---------------------------------------------------------------------------
// Packing kernels: build per-lane WMMA fragment layouts in workspace (bf16).
// A fragment (16x32, 16-bit): lane L<16 -> row L, elems {K0..7, K16..23};
//                             lane L>=16 -> row L-16, elems {K8..15, K24..31}.
// B fragment (32x16, 16-bit): lane L<16 -> col L, elems {K0..7, K16..23};
//                             lane L>=16 -> col L-16, elems {K8..15, K24..31}.
// ---------------------------------------------------------------------------

// x (B*T x N f32, row-major) -> A-fragments [mt][kt][lane][16]
__global__ void __launch_bounds__(256)
pack_x_kernel(const float* __restrict__ x, __bf16* __restrict__ xpk) {
  int gid = blockIdx.x * 256 + threadIdx.x;          // 1024*16*32 threads
  int lane = gid & 31;
  int kt   = (gid >> 5) & 15;
  int mt   = gid >> 9;
  int row  = lane & 15;
  int koff = (lane < 16) ? 0 : 8;
  const float* src = x + ((size_t)mt * 16 + row) * NN + kt * 32;
  __bf16* dst = xpk + (size_t)gid * 16;
#pragma unroll
  for (int e = 0; e < 8; ++e) dst[e]     = (__bf16)src[koff + e];
#pragma unroll
  for (int e = 0; e < 8; ++e) dst[8 + e] = (__bf16)src[koff + 16 + e];
}

// Wi (H x N f32) as B = Wi^T (N x H) -> [kt(16)][ct(64)][lane][16]
__global__ void __launch_bounds__(256)
pack_wi_kernel(const float* __restrict__ Wi, __bf16* __restrict__ wipk) {
  int gid = blockIdx.x * 256 + threadIdx.x;          // 16*64*32 threads
  int lane = gid & 31;
  int ct   = (gid >> 5) & 63;
  int kt   = gid >> 11;
  int col  = ct * 16 + (lane & 15);                  // h index
  int kb   = kt * 32 + ((lane < 16) ? 0 : 8);        // n index base
  __bf16* dst = wipk + (size_t)gid * 16;
#pragma unroll
  for (int e = 0; e < 16; ++e)
    dst[e] = (__bf16)Wi[(size_t)col * NN + kb + e + (e & 8)];
}

// Circulant C^T (H x H), CT[j][i] = w[(j - i + H/2) mod H] -> [kt(32)][ct(64)][lane][16]
__global__ void __launch_bounds__(256)
pack_ct_kernel(const float* __restrict__ w, __bf16* __restrict__ ctpk) {
  int gid = blockIdx.x * 256 + threadIdx.x;          // 32*64*32 threads
  int lane = gid & 31;
  int ct   = (gid >> 5) & 63;
  int kt   = gid >> 11;
  int col  = ct * 16 + (lane & 15);                  // i (output index)
  int kb   = kt * 32 + ((lane < 16) ? 0 : 8);        // j (input index) base
  __bf16* dst = ctpk + (size_t)gid * 16;
#pragma unroll
  for (int e = 0; e < 16; ++e) {
    int j = kb + e + (e & 8);
    int k = (j - col + (HH / 2)) & (HH - 1);
    dst[e] = (__bf16)w[k];
  }
}

// Wo (O x H f32) as B = Wo^T (H x O) -> [kt(32)][ct(32)][lane][16]
__global__ void __launch_bounds__(256)
pack_wo_kernel(const float* __restrict__ Wo, __bf16* __restrict__ wopk) {
  int gid = blockIdx.x * 256 + threadIdx.x;          // 32*32*32 threads
  int lane = gid & 31;
  int ct   = (gid >> 5) & 31;
  int kt   = gid >> 10;
  int col  = ct * 16 + (lane & 15);                  // o index
  int kb   = kt * 32 + ((lane < 16) ? 0 : 8);        // h index base
  __bf16* dst = wopk + (size_t)gid * 16;
#pragma unroll
  for (int e = 0; e < 16; ++e)
    dst[e] = (__bf16)Wo[(size_t)col * HH + kb + e + (e & 8)];
}

// ---------------------------------------------------------------------------
// GEMM1: x_proj[bt,h] = sum_n x[bt,n]*Wi[h,n] + bi[h]
// grid = 1024 blocks (one per 16-row strip), 256 threads = 8 waves, 8 ct/wave.
// ---------------------------------------------------------------------------
__global__ void __launch_bounds__(256)
xproj_gemm_kernel(const __bf16* __restrict__ xpk, const __bf16* __restrict__ wipk,
                  const float* __restrict__ bi, float* __restrict__ xproj) {
  int mt   = blockIdx.x;
  int wave = threadIdx.x >> 5;
  int lane = threadIdx.x & 31;
  v8f acc[8] = {};
  for (int kt = 0; kt < KT1; ++kt) {
    v16bf a = *(const v16bf*)(xpk + (((size_t)mt * KT1 + kt) * 32 + lane) * 16);
#pragma unroll
    for (int c = 0; c < 8; ++c) {
      int ct = wave * 8 + c;
      v16bf b = *(const v16bf*)(wipk + (((size_t)kt * CT1 + ct) * 32 + lane) * 16);
      acc[c] = __builtin_amdgcn_wmma_f32_16x16x32_bf16(
          false, a, false, b, (short)0, acc[c], false, false);
    }
  }
  int mrow = (lane < 16) ? 0 : 8;
  int ncol = lane & 15;
#pragma unroll
  for (int c = 0; c < 8; ++c) {
    int ct  = wave * 8 + c;
    int col = ct * 16 + ncol;
    float bv = bi[col];
#pragma unroll
    for (int r = 0; r < 8; ++r) {
      size_t row = (size_t)mt * 16 + mrow + r;
      xproj[row * HH + col] = acc[c][r] + bv;
    }
  }
}

// ---------------------------------------------------------------------------
// Recurrence: h = tanh(x_proj[:,t,:] + h @ C^T), z[:,t,:] = h.
// Independent per batch row -> 2 persistent workgroups of 16 batch rows each.
// h lives in LDS (16x1024 bf16 = 32KB); the x_proj tile of the current step is
// DMA'd into LDS by the Tensor Data Mover while WMMAs run on the current step.
// 32 waves, 2 column tiles per wave.
// ---------------------------------------------------------------------------
__global__ void __launch_bounds__(1024)
recurrence_kernel(const float* __restrict__ h0, const float* __restrict__ xproj,
                  const __bf16* __restrict__ ctpk, float* __restrict__ z) {
  __shared__ __align__(16) __bf16 hbuf[16 * HH];    // 32KB hidden state
  __shared__ __align__(16) float  xpbuf[16 * HH];   // 64KB TDM-landed xp tile
  int bb   = blockIdx.x;                    // batch rows [16*bb, 16*bb+16)
  int wave = threadIdx.x >> 5;
  int lane = threadIdx.x & 31;
  // init h from h0 (layout (1,B,H))
  for (int idx = threadIdx.x; idx < 16 * HH; idx += blockDim.x) {
    int r = idx >> 10, c = idx & (HH - 1);
    hbuf[idx] = (__bf16)h0[((size_t)(16 * bb + r)) * HH + c];
  }
  // TDM descriptor invariants for this block
  unsigned xp_lds   = (unsigned)(size_t)&xpbuf[0];
  unsigned long long xp_base =
      (unsigned long long)(const void*)(xproj + ((size_t)(16 * bb) * TT) * HH);
  if (wave == 0) {                          // prologue: fetch tile t=0
    tdm_load_xp_tile(xp_lds, xp_base);
  }
  __syncthreads();
  int mrow = (lane < 16) ? 0 : 8;
  int ncol = lane & 15;
  int koff = (lane < 16) ? 0 : 8;
  int arow = lane & 15;
  for (int t = 0; t < TT; ++t) {
    // ---- phase A: y = h @ C^T (reads hbuf; TDM for this t's xp in flight) ----
    v8f acc[2] = {};
    for (int kt = 0; kt < KTH; ++kt) {
      const __bf16* ap = &hbuf[arow * HH + kt * 32 + koff];
      v8bf lo = *(const v8bf*)ap;
      v8bf hi = *(const v8bf*)(ap + 16);
      v16bf a = cat16(lo, hi);
#pragma unroll
      for (int c = 0; c < 2; ++c) {
        int ct = wave * 2 + c;
        v16bf b = *(const v16bf*)(ctpk + (((size_t)kt * CTH + ct) * 32 + lane) * 16);
        acc[c] = __builtin_amdgcn_wmma_f32_16x16x32_bf16(
            false, a, false, b, (short)0, acc[c], false, false);
      }
    }
    if (wave == 0) __builtin_amdgcn_s_wait_tensorcnt((short)0);
    __syncthreads();   // hbuf reads done; xpbuf tile for step t landed
    // ---- phase B: h = tanh(xp + y); write z and hbuf ----
#pragma unroll
    for (int c = 0; c < 2; ++c) {
      int ct  = wave * 2 + c;
      int col = ct * 16 + ncol;
#pragma unroll
      for (int r = 0; r < 8; ++r) {
        int batch = 16 * bb + mrow + r;
        size_t bt = (size_t)batch * TT + t;
        float pre = acc[c][r] + xpbuf[(mrow + r) * HH + col];
        float hv  = tanhf(pre);
        z[bt * HH + col] = hv;
        hbuf[(mrow + r) * HH + col] = (__bf16)hv;
      }
    }
    __syncthreads();   // xpbuf reads + hbuf writes complete
    // ---- kick TDM for step t+1 (overlaps next phase A) ----
    if (wave == 0 && (t + 1) < TT) {
      tdm_load_xp_tile(xp_lds, xp_base + ((unsigned long long)(t + 1)) * HH * 4ull);
    }
  }
}

// ---------------------------------------------------------------------------
// GEMM3 + softmax: logits = z@Wo^T + bo, out = softmax(logits, axis=O).
// grid = 1024 blocks (16-row strip), 512 threads = 16 waves, 2 ct/wave.
// z strip staged to LDS as bf16; logits reuse the same LDS buffer as f32.
// ---------------------------------------------------------------------------
__global__ void __launch_bounds__(512)
out_gemm_softmax_kernel(const float* __restrict__ z, const __bf16* __restrict__ wopk,
                        const float* __restrict__ bo, float* __restrict__ out) {
  __shared__ __align__(16) __bf16 zbf[16 * HH];   // 32KB; reused as logits f32
  __shared__ float red[16 * 32];
  __shared__ float rowstat[16 * 2];
  float* lg = (float*)zbf;                        // 16*512 f32 = 32KB alias
  int mt   = blockIdx.x;
  int wave = threadIdx.x >> 5;
  int lane = threadIdx.x & 31;
  for (int idx = threadIdx.x; idx < 16 * HH; idx += blockDim.x) {
    int r = idx >> 10, c = idx & (HH - 1);
    zbf[idx] = (__bf16)z[((size_t)mt * 16 + r) * HH + c];
  }
  __syncthreads();
  int mrow = (lane < 16) ? 0 : 8;
  int ncol = lane & 15;
  int koff = (lane < 16) ? 0 : 8;
  int arow = lane & 15;
  v8f acc[2] = {};
  for (int kt = 0; kt < KTH; ++kt) {
    const __bf16* ap = &zbf[arow * HH + kt * 32 + koff];
    v8bf lo = *(const v8bf*)ap;
    v8bf hi = *(const v8bf*)(ap + 16);
    v16bf a = cat16(lo, hi);
#pragma unroll
    for (int c = 0; c < 2; ++c) {
      int ct = wave * 2 + c;
      v16bf b = *(const v16bf*)(wopk + (((size_t)kt * CTO + ct) * 32 + lane) * 16);
      acc[c] = __builtin_amdgcn_wmma_f32_16x16x32_bf16(
          false, a, false, b, (short)0, acc[c], false, false);
    }
  }
  __syncthreads();   // zbf reads done; safe to overwrite with logits
#pragma unroll
  for (int c = 0; c < 2; ++c) {
    int ct  = wave * 2 + c;
    int col = ct * 16 + ncol;
    float bv = bo[col];
#pragma unroll
    for (int r = 0; r < 8; ++r)
      lg[(mrow + r) * OO + col] = acc[c][r] + bv;
  }
  __syncthreads();
  // softmax: 512 threads; row = tid&15, seg = tid>>4 covers 16 cols each
  int row = threadIdx.x & 15;
  int seg = threadIdx.x >> 4;
  float m = -3.0e38f;
  for (int j = 0; j < 16; ++j) m = fmaxf(m, lg[row * OO + seg * 16 + j]);
  red[row * 32 + seg] = m;
  __syncthreads();
  if (threadIdx.x < 16) {
    float mm = red[threadIdx.x * 32];
    for (int s = 1; s < 32; ++s) mm = fmaxf(mm, red[threadIdx.x * 32 + s]);
    rowstat[threadIdx.x * 2] = mm;
  }
  __syncthreads();
  float rowmax = rowstat[row * 2];
  float ssum = 0.f;
  for (int j = 0; j < 16; ++j) ssum += expf(lg[row * OO + seg * 16 + j] - rowmax);
  red[row * 32 + seg] = ssum;
  __syncthreads();
  if (threadIdx.x < 16) {
    float ss = 0.f;
    for (int s = 0; s < 32; ++s) ss += red[threadIdx.x * 32 + s];
    rowstat[threadIdx.x * 2 + 1] = ss;
  }
  __syncthreads();
  float inv = 1.0f / rowstat[row * 2 + 1];
  for (int j = 0; j < 16; ++j) {
    int col = seg * 16 + j;
    out[((size_t)mt * 16 + row) * OO + col] =
        expf(lg[row * OO + col] - rowmax) * inv;
  }
}

// ---------------------------------------------------------------------------
extern "C" void kernel_launch(void* const* d_in, const int* in_sizes, int n_in,
                              void* d_out, int out_size, void* d_ws, size_t ws_size,
                              hipStream_t stream) {
  (void)in_sizes; (void)n_in; (void)out_size; (void)ws_size;
  const float* x  = (const float*)d_in[0];
  const float* h0 = (const float*)d_in[1];
  const float* Wi = (const float*)d_in[2];
  const float* bi = (const float*)d_in[3];
  const float* Wo = (const float*)d_in[4];
  const float* bo = (const float*)d_in[5];
  const float* w  = (const float*)d_in[6];

  float* out = (float*)d_out;                                // (B,T,O) softmax
  float* z   = (float*)d_out + (size_t)BB * TT * OO;         // (B,T,H) hidden seq

  // workspace layout (84 MB total)
  char* ws = (char*)d_ws;
  float*  xproj = (float*)ws;                                // 64 MB
  __bf16* xpk   = (__bf16*)(ws + ((size_t)64 << 20));        // 16 MB
  __bf16* wipk  = (__bf16*)(ws + ((size_t)80 << 20));        //  1 MB
  __bf16* ctpk  = (__bf16*)(ws + ((size_t)81 << 20));        //  2 MB
  __bf16* wopk  = (__bf16*)(ws + ((size_t)83 << 20));        //  1 MB

  // pack / convert
  pack_x_kernel <<<2048, 256, 0, stream>>>(x,  xpk);
  pack_wi_kernel<<<128,  256, 0, stream>>>(Wi, wipk);
  pack_ct_kernel<<<256,  256, 0, stream>>>(w,  ctpk);
  pack_wo_kernel<<<128,  256, 0, stream>>>(Wo, wopk);

  // GEMM1: x_proj
  xproj_gemm_kernel<<<MT1, 256, 0, stream>>>(xpk, wipk, bi, xproj);

  // persistent recurrence: 2 independent 16-batch-row blocks, TDM-fed
  recurrence_kernel<<<2, 1024, 0, stream>>>(h0, xproj, ctpk, z);

  // GEMM3 + softmax
  out_gemm_softmax_kernel<<<MT1, 512, 0, stream>>>(z, wopk, bo, out);
}
